// SimVectorQuantizer_14920716386721
// MI455X (gfx1250) — matrix operand order repaired and checked by reading, not compiled, not run on hardware
//
#include <hip/hip_runtime.h>

typedef __attribute__((ext_vector_type(16))) __bf16       v16bf;
typedef __attribute__((ext_vector_type(8)))  float        v8f;
typedef __attribute__((ext_vector_type(4)))  unsigned int u32x4;
typedef __attribute__((ext_vector_type(8)))  int          i32x8;
typedef __attribute__((ext_vector_type(4)))  int          i32x4;

#define BN   16384      // B*N
#define DD   256        // D
#define KCB  16384      // codebook size
#define BND  4194304    // B*N*D
#define KD   4194304    // K*D

// ---------------------------------------------------------------------------
// Kernel 1: emb = emb_weight @ proj_w^T + proj_b     (K x D) f32 out + bf16 ws
// One wave computes a 16x16 tile via 8x v_wmma_f32_16x16x32_bf16.
// ---------------------------------------------------------------------------
__global__ __launch_bounds__(256)
void proj_kernel(const float* __restrict__ embw,
                 const float* __restrict__ projw,
                 const float* __restrict__ projb,
                 float*  __restrict__ out_emb,
                 __bf16* __restrict__ embb)
{
    const int lane = threadIdx.x & 31;
    const int wave = threadIdx.x >> 5;
    const int tile = blockIdx.x * 8 + wave;      // 16384 tiles total
    const int rowBase = (tile >> 4) * 16;        // over K
    const int colBase = (tile & 15) * 16;        // over D

    const int mrow = lane & 15;
    const int hi   = lane >> 4;                  // 0: lanes 0-15, 1: lanes 16-31
    const int kA   = hi * 16;                    // A K-half (contiguous, matches B)
    const int kB   = hi * 16;                    // B K-half (32x16 B)
    const int col  = colBase + mrow;

    v8f acc = {};
    for (int c = 0; c < 8; ++c) {                // D=256 = 8 chunks of K=32
        const int d0 = c * 32;
        const float* ap = embw + (size_t)(rowBase + mrow) * DD + d0 + kA;
        v16bf a;
        #pragma unroll
        for (int i = 0; i < 8; ++i) { a[i] = (__bf16)ap[i]; a[8 + i] = (__bf16)ap[8 + i]; }
        // emb[k,n] = sum_d embw[k,d] * proj_w[n,d]  -> B[d][n] = proj_w[n][d] (contiguous)
        const float* bp = projw + (size_t)col * DD + d0 + kB;
        v16bf b;
        #pragma unroll
        for (int j = 0; j < 16; ++j) b[j] = (__bf16)bp[j];
        acc = __builtin_amdgcn_wmma_f32_16x16x32_bf16(false, a, false, b,
                                                      (short)0, acc, false, false);
    }
    const float bias = projb[col];
    #pragma unroll
    for (int v = 0; v < 8; ++v) {
        const int row = rowBase + hi * 8 + v;    // C/D layout: M = 8*hi + vgpr
        const float val = acc[v] + bias;
        out_emb[(size_t)row * DD + col] = val;
        embb[(size_t)row * DD + col]    = (__bf16)val;
    }
}

// ---------------------------------------------------------------------------
// Kernel 2: e2[k] = ||emb[k,:]||^2  (one wave per codebook row)
// ---------------------------------------------------------------------------
__global__ __launch_bounds__(256)
void e2_kernel(const float* __restrict__ emb, float* __restrict__ e2)
{
    const int lane = threadIdx.x & 31;
    const int row  = blockIdx.x * 8 + (threadIdx.x >> 5);
    const float* p = emb + (size_t)row * DD;
    float s = 0.f;
    #pragma unroll
    for (int j = 0; j < 8; ++j) { float v = p[lane + 32 * j]; s += v * v; }
    #pragma unroll
    for (int m = 16; m >= 1; m >>= 1) s += __shfl_xor(s, m, 32);
    if (lane == 0) e2[row] = s;
}

// ---------------------------------------------------------------------------
// Kernel 3: fused distance GEMM + argmin.
// Block = 256 threads (8 waves), owns 32 z-rows. z tile staged to LDS by the
// Tensor Data Mover; A held in registers (2 row-tiles); each wave sweeps 128
// codebook column tiles; argmin fused in the WMMA epilogue.
// ---------------------------------------------------------------------------
__global__ __launch_bounds__(256)
void dist_kernel(const float*  __restrict__ z,
                 const __bf16* __restrict__ embb,
                 const float*  __restrict__ e2,
                 int*          __restrict__ qidx)
{
    __shared__ float sA[32 * DD];          // 32 KB z tile
    __shared__ float sBv[8][32];
    __shared__ int   sBi[8][32];

    const int lane = threadIdx.x & 31;
    const int wave = threadIdx.x >> 5;
    const int row0 = blockIdx.x * 32;

    if (threadIdx.x < 32) {
        // TDM: DMA a 32x256 f32 tile of z into LDS (D# per ISA ch.8).
        unsigned long long ga = (unsigned long long)(uintptr_t)(z + (size_t)row0 * DD);
        unsigned int lds = (unsigned int)(uintptr_t)&sA[0];
        u32x4 g0 = { 1u,                                  // count=1, user load
                     lds,                                 // lds_addr [63:32]
                     (unsigned int)ga,                    // global_addr lo
                     (unsigned int)(ga >> 32) | (2u << 30) };   // addr hi | type=2
        i32x8 g1 = { 0x20000,                             // data_size=4B, mask=0
                     (int)(256u << 16),                   // tensor_dim0 = 256
                     (int)(16384u << 16),                 // tensor_dim1 = 16384
                     (int)(256u << 16),                   // tile_dim0 = 256
                     32,                                  // tile_dim1 = 32
                     256,                                 // tensor_dim0_stride = 256
                     0, 0 };
        i32x4 g2 = { 0, 0, 0, 0 };
        i32x4 g3 = { 0, 0, 0, 0 };
        i32x8 g4 = { 0, 0, 0, 0, 0, 0, 0, 0 };
        __builtin_amdgcn_tensor_load_to_lds(g0, g1, g2, g3, g4, 0);
        __builtin_amdgcn_s_wait_tensorcnt(0);
    }
    __syncthreads();

    const int mrow = lane & 15;
    const int hi   = lane >> 4;
    const int kA   = hi * 16;
    const int kB   = hi * 16;

    // Pre-convert both 16-row A tiles into WMMA register layout (held in VGPRs).
    v16bf a[2][8];
    #pragma unroll
    for (int t = 0; t < 2; ++t) {
        const float* rp = &sA[(t * 16 + mrow) * DD];
        #pragma unroll
        for (int c = 0; c < 8; ++c) {
            const int d0 = c * 32;
            #pragma unroll
            for (int i = 0; i < 8; ++i) {
                a[t][c][i]     = (__bf16)rp[d0 + kA + i];
                a[t][c][8 + i] = (__bf16)rp[d0 + 8 + kA + i];
            }
        }
    }

    float bv[2][8]; int bi[2][8];
    #pragma unroll
    for (int t = 0; t < 2; ++t)
        #pragma unroll
        for (int v = 0; v < 8; ++v) { bv[t][v] = 3.4e38f; bi[t][v] = 0; }

    // Sweep all 1024 codebook column tiles, 128 per wave; emb_bf16 is L2-resident.
    for (int ct = wave; ct < (KCB / 16); ct += 8) {
        const int col = ct * 16 + mrow;
        const __bf16* bp = embb + (size_t)col * DD + kB;
        const float e2c = e2[col];
        v8f acc0 = {}, acc1 = {};
        #pragma unroll
        for (int c = 0; c < 8; ++c) {
            v16bf b = *(const v16bf*)(bp + c * 32);    // one B load feeds 2 WMMAs
            acc0 = __builtin_amdgcn_wmma_f32_16x16x32_bf16(false, a[0][c], false, b,
                                                           (short)0, acc0, false, false);
            acc1 = __builtin_amdgcn_wmma_f32_16x16x32_bf16(false, a[1][c], false, b,
                                                           (short)0, acc1, false, false);
        }
        #pragma unroll
        for (int v = 0; v < 8; ++v) {
            float s0 = e2c - 2.0f * acc0[v];
            if (s0 < bv[0][v] || (s0 == bv[0][v] && col < bi[0][v])) { bv[0][v] = s0; bi[0][v] = col; }
            float s1 = e2c - 2.0f * acc1[v];
            if (s1 < bv[1][v] || (s1 == bv[1][v] && col < bi[1][v])) { bv[1][v] = s1; bi[1][v] = col; }
        }
    }

    // Reduce argmin across the 16 lanes of each half-wave (wave32, width 16).
    #pragma unroll
    for (int t = 0; t < 2; ++t)
        #pragma unroll
        for (int v = 0; v < 8; ++v) {
            float mv = bv[t][v]; int mi = bi[t][v];
            #pragma unroll
            for (int m = 8; m >= 1; m >>= 1) {
                float ov = __shfl_xor(mv, m, 16);
                int   oi = __shfl_xor(mi, m, 16);
                if (ov < mv || (ov == mv && oi < mi)) { mv = ov; mi = oi; }
            }
            bv[t][v] = mv; bi[t][v] = mi;
        }
    if (mrow == 0) {
        #pragma unroll
        for (int t = 0; t < 2; ++t)
            #pragma unroll
            for (int v = 0; v < 8; ++v) {
                const int r = t * 16 + hi * 8 + v;
                sBv[wave][r] = bv[t][v];
                sBi[wave][r] = bi[t][v];
            }
    }
    __syncthreads();
    if (threadIdx.x < 32) {
        float mv = sBv[0][threadIdx.x]; int mi = sBi[0][threadIdx.x];
        #pragma unroll
        for (int w = 1; w < 8; ++w) {
            float ov = sBv[w][threadIdx.x]; int oi = sBi[w][threadIdx.x];
            if (ov < mv || (ov == mv && oi < mi)) { mv = ov; mi = oi; }
        }
        qidx[row0 + threadIdx.x] = mi;
    }
}

// ---------------------------------------------------------------------------
// Kernel 4: outputs — z pass-through, quantized = emb[idx], idx as float.
// ---------------------------------------------------------------------------
__global__ __launch_bounds__(256)
void finalize_kernel(const float* __restrict__ z,
                     const float* __restrict__ emb,
                     const int*   __restrict__ qidx,
                     float* __restrict__ out_z,
                     float* __restrict__ out_q,
                     float* __restrict__ out_i)
{
    const int row = blockIdx.x;
    const int t   = threadIdx.x;
    const int idx = qidx[row];
    out_z[(size_t)row * DD + t] = z[(size_t)row * DD + t];
    out_q[(size_t)row * DD + t] = emb[(size_t)idx * DD + t];
    if (t == 0) out_i[row] = (float)idx;
}

extern "C" void kernel_launch(void* const* d_in, const int* in_sizes, int n_in,
                              void* d_out, int out_size, void* d_ws, size_t ws_size,
                              hipStream_t stream) {
    const float* z     = (const float*)d_in[0];   // (B,N,D)
    const float* embw  = (const float*)d_in[1];   // (K,D)
    const float* projw = (const float*)d_in[2];   // (D,D)
    const float* projb = (const float*)d_in[3];   // (D,)

    float* out   = (float*)d_out;
    float* out_z = out;                  // BND
    float* out_e = out + BND;            // KD
    float* out_q = out + BND + KD;       // BND
    float* out_i = out + 2 * BND + KD;   // BN

    __bf16* embb = (__bf16*)d_ws;                                   // 8 MB
    float*  e2   = (float*)((char*)d_ws + (size_t)KD * 2);          // 64 KB
    int*    qidx = (int*)((char*)d_ws + (size_t)KD * 2 + 65536);    // 64 KB

    proj_kernel    <<<2048,  256, 0, stream>>>(embw, projw, projb, out_e, embb);
    e2_kernel      <<<2048,  256, 0, stream>>>(out_e, e2);
    dist_kernel    <<<512,   256, 0, stream>>>(z, embb, e2, qidx);
    finalize_kernel<<<BN,    256, 0, stream>>>(z, out_e, qidx, out_z, out_q, out_i);
}
